// CohereAttention_86758339379759
// MI455X (gfx1250) — matrix operand order repaired and checked by reading, not compiled, not run on hardware
//
#include <hip/hip_runtime.h>

// ---------------- problem constants ----------------
#define HIDDEN 4096
#define NH     32
#define NKV    8
#define HD     128
#define QS     (NH * HD)            // 4096
#define KVS    (NKV * HD)           // 1024
#define QKV_N  (QS + 2 * KVS)       // 6144
#define SEQ    2048
#define BB     2
#define NTOK   (BB * SEQ)           // 4096
#define EPSV   1e-5f
#define SCALE_V 0.08838834764831845f   // 128^-0.5
#define LN_THETA 9.210340371976184f    // ln(10000)

typedef _Float16 h16 __attribute__((ext_vector_type(16)));
typedef _Float16 h4  __attribute__((ext_vector_type(4)));
typedef float    f32x8 __attribute__((ext_vector_type(8)));
typedef unsigned int uintx4 __attribute__((ext_vector_type(4)));
typedef int intx4 __attribute__((ext_vector_type(4)));
typedef int intx8 __attribute__((ext_vector_type(8)));

union FragU {
  uint4 u[2];
  h16   v;
};

__device__ inline _Float16 f2h(float f) { return (_Float16)f; }

__device__ inline f32x8 zero8() {
  f32x8 z = {0.f, 0.f, 0.f, 0.f, 0.f, 0.f, 0.f, 0.f};
  return z;
}

// reduce across all 32 lanes (wave32)
__device__ inline float wave_sum32(float v) {
#pragma unroll
  for (int m = 1; m < 32; m <<= 1) v += __shfl_xor(v, m, 32);
  return v;
}
// reduce within each half-wave group of 16 lanes
__device__ inline float half_max16(float v) {
#pragma unroll
  for (int m = 1; m < 16; m <<= 1) v = fmaxf(v, __shfl_xor(v, m, 32));
  return v;
}
__device__ inline float half_sum16(float v) {
#pragma unroll
  for (int m = 1; m < 16; m <<= 1) v += __shfl_xor(v, m, 32);
  return v;
}

// ---------------- Tensor Data Mover: 2D tile -> LDS ----------------
// Builds the D# descriptor (CDNA5 ISA) for a 2D tile of
// 16-bit elements and issues TENSOR_LOAD_TO_LDS (clang-23 6-arg builtin form).
// pad_interval code c -> pad every 2^(c+1) DWORDs; pad_amount code a -> insert
// (a+1) DWORDs. Caller must be wave-uniform; completion via s_wait_tensorcnt.
__device__ inline void tdm_load_2d(unsigned ldsOff, const void* gptr,
                                   unsigned tensor_d0, unsigned tensor_d1,
                                   long long stride0,
                                   unsigned tile_d0, unsigned tile_d1,
                                   unsigned pad_interval_code,
                                   unsigned pad_amount_code) {
  unsigned long long ga = (unsigned long long)(uintptr_t)gptr;
  uintx4 g0;
  g0.x = 1u;                                             // count = 1 (valid D#)
  g0.y = ldsOff;                                         // lds_addr (bytes)
  g0.z = (unsigned)(ga & 0xFFFFFFFFull);                 // global_addr[31:0]
  g0.w = (unsigned)((ga >> 32) & 0x01FFFFFFull) | (2u << 30);  // addr[56:32] | type=2
  intx8 g1;
  unsigned d0 = (1u << 16)                               // data_size = 2 bytes
              | (1u << 20)                               // pad_enable
              | (pad_interval_code << 22)
              | (pad_amount_code << 25);
  g1[0] = (int)d0;                                       // workgroup_mask = 0
  g1[1] = (int)((tensor_d0 & 0xFFFFu) << 16);            // dim0[15:0] (barrier addr 0)
  g1[2] = (int)(((tensor_d0 >> 16) & 0xFFFFu) | ((tensor_d1 & 0xFFFFu) << 16));
  g1[3] = (int)(((tensor_d1 >> 16) & 0xFFFFu) | ((tile_d0 & 0xFFFFu) << 16));
  g1[4] = (int)(tile_d1 & 0xFFFFu);                      // tile_dim2 = 0
  g1[5] = (int)(unsigned)(stride0 & 0xFFFFFFFFll);       // dim0 stride lo
  g1[6] = (int)(unsigned)((stride0 >> 32) & 0xFFFFll);   // stride hi16 | d1stride lo = 0
  g1[7] = 0;
  intx4 gz4 = {0, 0, 0, 0};                              // 2D: groups 2/3 unused
  intx8 gz8 = {0, 0, 0, 0, 0, 0, 0, 0};
  __builtin_amdgcn_tensor_load_to_lds(g0, g1, gz4, gz4, gz8, 0);
}

// ---------------- kernel 0: f32 -> f16 convert ----------------
__global__ __launch_bounds__(256)
void cvt_f32_f16(const float* __restrict__ src, _Float16* __restrict__ dst, long n) {
  long i = ((long)blockIdx.x * blockDim.x + threadIdx.x) * 4;
  if (i >= n) return;
  float4 x = *reinterpret_cast<const float4*>(src + i);
  h4 o;
  o.x = f2h(x.x); o.y = f2h(x.y); o.z = f2h(x.z); o.w = f2h(x.w);
  *reinterpret_cast<h4*>(dst + i) = o;
}

// ---------------- kernel 1/4: tiled WMMA GEMM  C[M,N] = A[M,K] * B[K,N] ----------------
#define BM 128
#define BN 128
#define BK 32
#define LDK 40   // padded LDS row length (80B rows keep 16B alignment, kill conflicts)

__global__ __launch_bounds__(256)
void gemm_f16_f32(const _Float16* __restrict__ A,
                  const _Float16* __restrict__ Bm,
                  float* __restrict__ C,
                  int M, int N, int K) {
  __shared__ __align__(16) _Float16 ldsA[BM][LDK];   // [row][k]  (TDM-filled, padded)
  __shared__ __align__(16) _Float16 ldsBt[BN][LDK];  // [n][k]    (transposed by threads)

  const int tid  = threadIdx.x;
  const int lane = tid & 31;
  const int wave = tid >> 5;
  const int wm   = wave & 3;   // 4 wave-rows of 32
  const int wn   = wave >> 2;  // 2 wave-cols of 64
  const int m0 = blockIdx.y * BM;
  const int n0 = blockIdx.x * BN;
  const int r16 = lane & 15;
  const int hlf = lane >> 4;

  const unsigned ldsAoff = (unsigned)(uintptr_t)&ldsA[0][0];

  f32x8 acc[2][4];
#pragma unroll
  for (int i = 0; i < 2; i++)
#pragma unroll
    for (int j = 0; j < 4; j++) acc[i][j] = zero8();

  const int brow = tid >> 3;         // k 0..31
  const int bseg = (tid & 7) * 16;   // n 0..112

  for (int k0 = 0; k0 < K; k0 += BK) {
    // B tile: 32B coalesced global load per thread (transposed into LDS below)
    const _Float16* gb = Bm + (size_t)(k0 + brow) * N + n0 + bseg;
    union { uint4 q; unsigned short s[8]; } b0, b1;
    b0.q = *reinterpret_cast<const uint4*>(gb);
    b1.q = *reinterpret_cast<const uint4*>(gb + 8);
    if (k0 + BK < K)
      __builtin_prefetch(Bm + (size_t)(k0 + BK + brow) * N + n0 + bseg, 0, 1);

    __syncthreads();  // all prior LDS reads consumed (DScnt==0 at barrier)

    // A tile via Tensor Data Mover: 128 rows x 32 f16, pad 16B per 64B row -> LDK=40
    if (wave == 0)
      tdm_load_2d(ldsAoff, A + (size_t)m0 * K + k0,
                  (unsigned)K, (unsigned)M, (long long)K,
                  BK, BM, /*interval 16 DW*/3, /*pad 4 DW*/3);

    unsigned short* lb = reinterpret_cast<unsigned short*>(&ldsBt[0][0]);
#pragma unroll
    for (int i = 0; i < 8; i++) lb[(bseg + i) * LDK + brow]     = b0.s[i];
#pragma unroll
    for (int i = 0; i < 8; i++) lb[(bseg + 8 + i) * LDK + brow] = b1.s[i];

    if (wave == 0) __builtin_amdgcn_s_wait_tensorcnt(0);
    __syncthreads();

    // fragments (16-bit A 16x32 / B 32x16, contiguous k: lane half h holds k = 16h..16h+15)
    FragU afr[2], bfr[4];
#pragma unroll
    for (int i = 0; i < 2; i++) {
      const _Float16* p = &ldsA[wm * 32 + i * 16 + r16][hlf * 16];
      afr[i].u[0] = *reinterpret_cast<const uint4*>(p);
      afr[i].u[1] = *reinterpret_cast<const uint4*>(p + 8);  // K = 8 + hlf*16
    }
#pragma unroll
    for (int j = 0; j < 4; j++) {
      const _Float16* p = &ldsBt[wn * 64 + j * 16 + r16][hlf * 16];
      bfr[j].u[0] = *reinterpret_cast<const uint4*>(p);
      bfr[j].u[1] = *reinterpret_cast<const uint4*>(p + 8);
    }
#pragma unroll
    for (int i = 0; i < 2; i++)
#pragma unroll
      for (int j = 0; j < 4; j++)
        acc[i][j] = __builtin_amdgcn_wmma_f32_16x16x32_f16(
            false, afr[i].v, false, bfr[j].v, (short)0, acc[i][j], false, false);
  }

  // C layout: VGPR jj -> row jj + 8*hlf, col = lane&15
#pragma unroll
  for (int i = 0; i < 2; i++) {
#pragma unroll
    for (int j = 0; j < 4; j++) {
      int col = n0 + wn * 64 + j * 16 + r16;
#pragma unroll
      for (int jj = 0; jj < 8; jj++) {
        int row = m0 + wm * 32 + i * 16 + hlf * 8 + jj;
        C[(size_t)row * N + col] = acc[i][j][jj];
      }
    }
  }
}

// ---------------- kernel 2: head-wise LN + GPT-J RoPE + split/transpose ----------------
__global__ __launch_bounds__(128)
void ln_rope_split(const float* __restrict__ qkv,
                   const int* __restrict__ positions,
                   const float* __restrict__ qw,
                   const float* __restrict__ kw,
                   _Float16* __restrict__ Qo,   // [B,NH,S,D]
                   _Float16* __restrict__ Ko,   // [B,NKV,S,D]
                   _Float16* __restrict__ Vt) { // [B,NKV,D,S]
  const int blk   = blockIdx.x;
  const int token = blk / 12;
  const int hb    = blk % 12;
  const int wave  = threadIdx.x >> 5;
  const int lane  = threadIdx.x & 31;
  const int head  = hb * 4 + wave;          // 0..47
  const int b = token >> 11;                // /SEQ
  const int s = token & (SEQ - 1);

  int col;
  if (head < NH)            col = head * HD;
  else if (head < NH + NKV) col = QS + (head - NH) * HD;
  else                      col = QS + KVS + (head - NH - NKV) * HD;

  float4 x = *reinterpret_cast<const float4*>(qkv + (size_t)token * QKV_N + col + lane * 4);

  if (head >= NH + NKV) {  // V: convert + store transposed [d][s]
    int kv = head - NH - NKV;
    size_t base = (size_t)(b * NKV + kv) * HD * SEQ + s;
    int d0 = lane * 4;
    Vt[base + (size_t)(d0 + 0) * SEQ] = f2h(x.x);
    Vt[base + (size_t)(d0 + 1) * SEQ] = f2h(x.y);
    Vt[base + (size_t)(d0 + 2) * SEQ] = f2h(x.z);
    Vt[base + (size_t)(d0 + 3) * SEQ] = f2h(x.w);
    return;
  }

  // head-wise layernorm over 128 elems (4 per lane)
  float mean = wave_sum32(x.x + x.y + x.z + x.w) * (1.0f / HD);
  float d0 = x.x - mean, d1 = x.y - mean, d2 = x.z - mean, d3 = x.w - mean;
  float var = wave_sum32(d0 * d0 + d1 * d1 + d2 * d2 + d3 * d3) * (1.0f / HD);
  float inv = rsqrtf(var + EPSV);
  const float* wp = (head < NH) ? (qw + head * HD) : (kw + (head - NH) * HD);
  float4 wv = *reinterpret_cast<const float4*>(wp + lane * 4);
  float y0 = d0 * inv * wv.x, y1 = d1 * inv * wv.y;
  float y2 = d2 * inv * wv.z, y3 = d3 * inv * wv.w;

  // GPT-J RoPE: lane holds pairs p0=2*lane, p1=2*lane+1
  float pos = (float)positions[token];
  float e0 = (float)(2 * lane) * (-2.0f / HD) * LN_THETA;
  float e1 = (float)(2 * lane + 1) * (-2.0f / HD) * LN_THETA;
  float a0 = pos * __expf(e0), a1 = pos * __expf(e1);
  float s0, c0, s1, c1;
  __sincosf(a0, &s0, &c0);
  __sincosf(a1, &s1, &c1);
  float o0 = y0 * c0 - y1 * s0;
  float o1 = y1 * c0 + y0 * s0;
  float o2 = y2 * c1 - y3 * s1;
  float o3 = y3 * c1 + y2 * s1;

  h4 out;
  out.x = f2h(o0); out.y = f2h(o1); out.z = f2h(o2); out.w = f2h(o3);
  _Float16* dst;
  if (head < NH) dst = Qo + ((size_t)(b * NH + head) * SEQ + s) * HD + lane * 4;
  else           dst = Ko + ((size_t)(b * NKV + (head - NH)) * SEQ + s) * HD + lane * 4;
  *reinterpret_cast<h4*>(dst) = out;
}

// ---------------- kernel 3: flash attention (causal, GQA 4:1) ----------------
// block = 128 q rows (8 waves x 16 rows). Shared K/V tiles are TDM-loaded into
// LDS once per 32 keys (8x traffic reduction vs per-wave loads); waves past
// their causal bound skip compute (wave-uniform) but keep hitting barriers.
#define KLD 136   // 128 dims + 8 pad (272B rows)
#define VLD 40    // 32 tokens + 8 pad (80B rows)

__global__ __launch_bounds__(256)
void flash_attn(const _Float16* __restrict__ Q,   // [B,NH,S,D]
                const _Float16* __restrict__ Kg,  // [B,NKV,S,D]
                const _Float16* __restrict__ Vt,  // [B,NKV,D,S]
                _Float16* __restrict__ attn) {    // [B,S,NH*D]
  __shared__ __align__(16) _Float16 ldsK[32][KLD];     // [token][dim]
  __shared__ __align__(16) _Float16 ldsV[HD][VLD];     // [dim][token]
  __shared__ __align__(16) _Float16 ldsP[8][16][40];   // per-wave P relayout

  const int qt = blockIdx.x, h = blockIdx.y, b = blockIdx.z;
  const int kv   = h >> 2;
  const int wave = threadIdx.x >> 5;
  const int lane = threadIdx.x & 31;
  const int r16  = lane & 15;
  const int hlf  = lane >> 4;
  const int qBase = qt * 128 + wave * 16;

  const _Float16* Qp = Q  + (size_t)(b * NH  + h)  * SEQ * HD;
  const _Float16* Kp = Kg + (size_t)(b * NKV + kv) * SEQ * HD;
  const _Float16* Vp = Vt + (size_t)(b * NKV + kv) * HD * SEQ;
  const unsigned ldsKoff = (unsigned)(uintptr_t)&ldsK[0][0];
  const unsigned ldsVoff = (unsigned)(uintptr_t)&ldsV[0][0];

  // Q fragments: 16 rows x 128 dims = 4 A-frags of 16x32
  FragU qf[4];
#pragma unroll
  for (int f = 0; f < 4; f++) {
    const _Float16* p = Qp + (size_t)(qBase + r16) * HD + f * 32 + hlf * 16;
    qf[f].u[0] = *reinterpret_cast<const uint4*>(p);
    qf[f].u[1] = *reinterpret_cast<const uint4*>(p + 8);
  }

  f32x8 o[8];
#pragma unroll
  for (int v = 0; v < 8; v++) o[v] = zero8();
  float mrun[8], lrun[8];
#pragma unroll
  for (int jj = 0; jj < 8; jj++) { mrun[jj] = -1e30f; lrun[jj] = 0.f; }

  const int myIter    = ((qBase + 15) >> 5) + 1;        // this wave's causal bound
  const int blockIter = ((qt * 128 + 127) >> 5) + 1;    // block-wide bound

  for (int it = 0; it < blockIter; ++it) {
    const int k32 = it * 32;

    __syncthreads();  // prior tile reads consumed (DScnt==0 per wave at barrier)
    if (wave == 0) {
      // K tile: 32 tokens x 128 dims (256B rows, +16B pad -> KLD=136)
      tdm_load_2d(ldsKoff, Kp + (size_t)k32 * HD,
                  HD, SEQ, (long long)HD, HD, 32, /*64 DW*/5, /*4 DW*/3);
      // V tile: 128 dims x 32 tokens from V^T (64B rows, +16B pad -> VLD=40)
      tdm_load_2d(ldsVoff, Vp + k32,
                  SEQ, HD, (long long)SEQ, 32, HD, /*16 DW*/3, /*4 DW*/3);
      __builtin_amdgcn_s_wait_tensorcnt(0);
    }
    __syncthreads();

    if (it < myIter) {  // wave-uniform: EXEC stays all-ones inside
      // scores: two 16x16 C tiles over 32 keys (K-dim 128 = 4 wmma steps each)
      f32x8 sc[2];
      sc[0] = zero8(); sc[1] = zero8();
#pragma unroll
      for (int t = 0; t < 2; t++) {
#pragma unroll
        for (int ks = 0; ks < 4; ks++) {
          FragU kf;  // B-frag: lane = key token, elems = contiguous head dims
          const _Float16* p = &ldsK[t * 16 + r16][ks * 32 + hlf * 16];
          kf.u[0] = *reinterpret_cast<const uint4*>(p);
          kf.u[1] = *reinterpret_cast<const uint4*>(p + 8);
          sc[t] = __builtin_amdgcn_wmma_f32_16x16x32_f16(
              false, qf[ks].v, false, kf.v, (short)0, sc[t], false, false);
        }
      }

      // scale + causal mask + online softmax (row = jj + 8*hlf, col = r16)
      float mt[8];
#pragma unroll
      for (int jj = 0; jj < 8; jj++) {
        int qrow = qBase + hlf * 8 + jj;
        float s0 = sc[0][jj] * SCALE_V;
        float s1 = sc[1][jj] * SCALE_V;
        if (k32 + r16 > qrow)      s0 = -1e30f;
        if (k32 + 16 + r16 > qrow) s1 = -1e30f;
        sc[0][jj] = s0; sc[1][jj] = s1;
        mt[jj] = half_max16(fmaxf(s0, s1));
      }
#pragma unroll
      for (int jj = 0; jj < 8; jj++) {
        float mnew = fmaxf(mrun[jj], mt[jj]);
        float corr = __expf(mrun[jj] - mnew);
        float p0 = __expf(sc[0][jj] - mnew);
        float p1 = __expf(sc[1][jj] - mnew);
        lrun[jj] = lrun[jj] * corr + half_sum16(p0 + p1);
        mrun[jj] = mnew;
#pragma unroll
        for (int v = 0; v < 8; v++) o[v][jj] *= corr;
        ldsP[wave][hlf * 8 + jj][r16]      = f2h(p0);
        ldsP[wave][hlf * 8 + jj][16 + r16] = f2h(p1);
      }

      // reload P as A-frag (16x32); per-wave LDS, DS in-order within wave
      FragU pf;
      const _Float16* pp = &ldsP[wave][r16][hlf * 16];
      pf.u[0] = *reinterpret_cast<const uint4*>(pp);
      pf.u[1] = *reinterpret_cast<const uint4*>(pp + 8);

      // O += P * V  (B-frag: lane = out dim, elems = contiguous tokens)
#pragma unroll
      for (int v = 0; v < 8; v++) {
        FragU vf;
        const _Float16* p = &ldsV[v * 16 + r16][hlf * 16];
        vf.u[0] = *reinterpret_cast<const uint4*>(p);
        vf.u[1] = *reinterpret_cast<const uint4*>(p + 8);
        o[v] = __builtin_amdgcn_wmma_f32_16x16x32_f16(
            false, pf.v, false, vf.v, (short)0, o[v], false, false);
      }
    }
  }

  // epilogue: O /= l, write attn [b, s, h*128 + d] in f16
#pragma unroll
  for (int v = 0; v < 8; v++) {
    int col = h * HD + v * 16 + r16;
#pragma unroll
    for (int jj = 0; jj < 8; jj++) {
      int srow = qBase + hlf * 8 + jj;
      attn[((size_t)(b * SEQ + srow)) * QS + col] = f2h(o[v][jj] / lrun[jj]);
    }
  }
}

// ---------------- host launch ----------------
extern "C" void kernel_launch(void* const* d_in, const int* in_sizes, int n_in,
                              void* d_out, int out_size, void* d_ws, size_t ws_size,
                              hipStream_t stream) {
  const int*   positions = (const int*)  d_in[0];
  const float* hidden    = (const float*)d_in[1];
  const float* Wqkv      = (const float*)d_in[2];
  const float* qw        = (const float*)d_in[3];
  const float* kw        = (const float*)d_in[4];
  const float* Wo        = (const float*)d_in[5];
  float* out = (float*)d_out;

  char* ws = (char*)d_ws;
  size_t off = 0;
  auto carve = [&](size_t bytes) -> char* {
    char* p = ws + off;
    off += (bytes + 255) & ~(size_t)255;
    return p;
  };
  _Float16* hH   = (_Float16*)carve((size_t)NTOK * HIDDEN * 2);
  _Float16* wqH  = (_Float16*)carve((size_t)HIDDEN * QKV_N * 2);
  _Float16* woH  = (_Float16*)carve((size_t)QS * HIDDEN * 2);
  float*    qkvF = (float*)   carve((size_t)NTOK * QKV_N * 4);
  _Float16* Qh   = (_Float16*)carve((size_t)BB * NH  * SEQ * HD * 2);
  _Float16* Kh   = (_Float16*)carve((size_t)BB * NKV * SEQ * HD * 2);
  _Float16* Vth  = (_Float16*)carve((size_t)BB * NKV * HD * SEQ * 2);
  _Float16* attn = (_Float16*)carve((size_t)NTOK * QS * 2);
  (void)in_sizes; (void)n_in; (void)out_size; (void)ws_size;

  long nh = (long)NTOK * HIDDEN, nw = (long)HIDDEN * QKV_N, no = (long)QS * HIDDEN;
  cvt_f32_f16<<<(int)((nh / 4 + 255) / 256), 256, 0, stream>>>(hidden, hH, nh);
  cvt_f32_f16<<<(int)((nw / 4 + 255) / 256), 256, 0, stream>>>(Wqkv, wqH, nw);
  cvt_f32_f16<<<(int)((no / 4 + 255) / 256), 256, 0, stream>>>(Wo, woH, no);

  gemm_f16_f32<<<dim3(QKV_N / BN, NTOK / BM), 256, 0, stream>>>(
      hH, wqH, qkvF, NTOK, QKV_N, HIDDEN);

  ln_rope_split<<<NTOK * 12, 128, 0, stream>>>(qkvF, positions, qw, kw, Qh, Kh, Vth);

  flash_attn<<<dim3(SEQ / 128, NH, BB), 256, 0, stream>>>(Qh, Kh, Vth, attn);

  gemm_f16_f32<<<dim3(HIDDEN / BN, NTOK / BM), 256, 0, stream>>>(
      attn, woH, out, NTOK, HIDDEN, QS);
}
